// GraphEncoder_9672266350628
// MI455X (gfx1250) — compile-verified, matrix operand order checked by reading, not run on hardware
//
#include <hip/hip_runtime.h>
#include <hip/hip_bf16.h>

#define N_NODES 50000
#define N_EDGES 200000
#define IN_DIM  16
#define NODE_DIM 64
#define EDGE_DIM 16
#define HIDDEN  256
#define ENN_H1  256
#define ENN_H2  1024
#define WE_DIM  1024   // IN_DIM * NODE_DIM

typedef __attribute__((ext_vector_type(16))) __bf16 v16bf;
typedef __attribute__((ext_vector_type(8)))  __bf16 v8bf;
typedef __attribute__((ext_vector_type(8)))  float  v8f;

__device__ __forceinline__ float elu1(float v) {
    return v > 0.f ? v : (__expf(v) - 1.f);
}

// K index inside a 16-bit A/B fragment element slot (ISA 7.12.2 16-bit layout):
// element e: K = (e/8)*16 + g*8 + (e%8)   (g = lane/16) -- two contiguous runs of 8
__device__ __forceinline__ int frag_k(int e, int g) {
    int j = e >> 1;
    return ((j >> 2) << 4) | (g << 3) | ((j & 3) << 1) | (e & 1);
}

__device__ __forceinline__ v16bf join8(v8bf a, v8bf b) {
    return __builtin_shufflevector(a, b, 0, 1, 2, 3, 4, 5, 6, 7,
                                   8, 9, 10, 11, 12, 13, 14, 15);
}

// A fragment (16xK row-major bf16 tile in LDS): per lane = 2 contiguous 16B runs
__device__ __forceinline__ v16bf load_a_lds(const __bf16* s, int lda, int koff, int lane) {
    int m = lane & 15, g = lane >> 4;
    const __bf16* row = s + m * lda + koff + g * 8;
    v8bf lo = *(const v8bf*)(row);
    v8bf hi = *(const v8bf*)(row + 16);
    return join8(lo, hi);
}

// B fragment from pre-packed fragment-major weight block (32 lanes x 16 bf16 = 1KB)
__device__ __forceinline__ v16bf load_b_pk(const __bf16* block, int lane) {
    const __bf16* p = block + lane * 16;
    v8bf lo = *(const v8bf*)(p);
    v8bf hi = *(const v8bf*)(p + 8);
    return join8(lo, hi);
}

// A fragment straight from global f32 row-major [rows x lda], K-guarded (zero pad)
__device__ __forceinline__ v16bf load_a_gl(const float* A, long row0, int lda,
                                           int koff, int kmax, int lane) {
    int m = lane & 15, g = lane >> 4;
    v16bf a;
#pragma unroll
    for (int e = 0; e < 16; ++e) {
        int k = koff + frag_k(e, g);
        a[e] = (__bf16)((k < kmax) ? A[(row0 + m) * (long)lda + k] : 0.f);
    }
    return a;
}

__device__ __forceinline__ v8f wmma_bf16(v16bf a, v16bf b, v8f c) {
    return __builtin_amdgcn_wmma_f32_16x16x32_bf16(false, a, false, b,
                                                   (short)0, c, false, false);
}

__device__ __forceinline__ void gatomAdd(float* p, float v) {
    unsafeAtomicAdd(p, v);   // global_atomic_add_f32
}

// ---------------------------------------------------------------------------
__global__ void k_zero(float* __restrict__ p, int n) {
    int i = blockIdx.x * blockDim.x + threadIdx.x;
    if (i < n) p[i] = 0.f;
}

// ---------------------------------------------------------------------------
// Pack f32 weight [K x N] into fragment-major bf16 blocks:
// block(kt,nt) at P[(kt*Nt+nt)*512], element [lane*16 + e] = W[kt*32+frag_k(e,g)][nt*16+n]
// K zero-padded to Kt*32.
__global__ void k_pack(const float* __restrict__ W, __bf16* __restrict__ P,
                       int Kreal, int Nt, int N, int total) {
    int idx = blockIdx.x * blockDim.x + threadIdx.x;
    if (idx >= total) return;
    int blk = idx >> 9, within = idx & 511;
    int lane = within >> 4, e = within & 15;
    int kt = blk / Nt, nt = blk - kt * Nt;
    int g = lane >> 4, n = lane & 15;
    int k = kt * 32 + frag_k(e, g);
    float v = (k < Kreal) ? W[(long)k * N + nt * 16 + n] : 0.f;
    P[idx] = (__bf16)v;
}

// ---------------------------------------------------------------------------
// Fused NNConv edge pipeline: 16 edges / workgroup, 4 waves.
// attr(16x16) -> h1(16x256) -> h2(16x1024) -> w_e(16x1024) fused into msg(16x64)
__global__ __launch_bounds__(128) void k_edge(
    const float* __restrict__ x, const int* __restrict__ ei,
    const float* __restrict__ eattr,
    const __bf16* __restrict__ pw1, const float* __restrict__ b1,
    const __bf16* __restrict__ pw2, const float* __restrict__ b2,
    const __bf16* __restrict__ pw3, const float* __restrict__ b3,
    float* __restrict__ agg) {
    __shared__ __bf16 s_attr[16][32];    // K zero-padded 16->32
    __shared__ __bf16 s_h1[16][256];
    __shared__ __bf16 s_h2[16][1024];    // 32 KB
    __shared__ float  s_x[16][16];       // gathered x[src]
    __shared__ float  s_msg[16][64];
    __shared__ int    s_dst[16];

    int tid = threadIdx.x, wave = tid >> 5, lane = tid & 31;
    long e0 = (long)blockIdx.x * 16;
    int g = lane >> 4, n = lane & 15;

    for (int i = tid; i < 16 * 64; i += 128) (&s_msg[0][0])[i] = 0.f;
    for (int i = tid; i < 16 * 32; i += 128) {
        int m = i >> 5, k = i & 31;
        s_attr[m][k] = (__bf16)((k < EDGE_DIM) ? eattr[(e0 + m) * EDGE_DIM + k] : 0.f);
    }
    if (tid < 16) s_dst[tid] = ei[N_EDGES + e0 + tid];
    for (int i = tid; i < 16 * 16; i += 128) {
        int m = i >> 4, k = i & 15;
        long s = ei[e0 + m];
        s_x[m][k] = x[s * IN_DIM + k];
    }
    __syncthreads();

    // --- enn layer 1: [16,16(pad32)] @ [16,256] ---
    {
        v16bf a = load_a_lds(&s_attr[0][0], 32, 0, lane);
        for (int t = wave; t < 16; t += 4) {
            v8f acc = {};
            acc = wmma_bf16(a, load_b_pk(pw1 + t * 512, lane), acc);
            int col = t * 16 + n;
#pragma unroll
            for (int r = 0; r < 8; ++r)
                s_h1[g * 8 + r][col] = (__bf16)elu1(acc[r] + b1[col]);
        }
    }
    __syncthreads();

    // --- enn layer 2: [16,256] @ [256,1024] ---
    {
        v16bf af[8];
#pragma unroll
        for (int ks = 0; ks < 8; ++ks) af[ks] = load_a_lds(&s_h1[0][0], 256, ks * 32, lane);
        for (int t = wave; t < 64; t += 4) {
            v8f acc = {};
#pragma unroll
            for (int ks = 0; ks < 8; ++ks)
                acc = wmma_bf16(af[ks], load_b_pk(pw2 + (ks * 64 + t) * 512, lane), acc);
            int col = t * 16 + n;
#pragma unroll
            for (int r = 0; r < 8; ++r)
                s_h2[g * 8 + r][col] = (__bf16)elu1(acc[r] + b2[col]);
        }
    }
    __syncthreads();

    // --- enn layer 3 fused with msg: w_e = ELU([16,1024]@[1024,1024]+b3),
    //     msg[e,o] += x[src[e],i] * w_e[e, i*64+o]  (col = i*64+o) ---
    {
        v8f zero = {};
        v8f acc[16];
#pragma unroll
        for (int ti = 0; ti < 16; ++ti) acc[ti] = zero;
        for (int ks = 0; ks < 32; ++ks) {
            v16bf a = load_a_lds(&s_h2[0][0], 1024, ks * 32, lane);
#pragma unroll
            for (int ti = 0; ti < 16; ++ti) {
                int t = wave + ti * 4;
                acc[ti] = wmma_bf16(a, load_b_pk(pw3 + (ks * 64 + t) * 512, lane), acc[ti]);
            }
        }
        for (int ti = 0; ti < 16; ++ti) {
            int col = (wave + ti * 4) * 16 + n;
            int i = col >> 6, o = col & 63;
#pragma unroll
            for (int r = 0; r < 8; ++r) {
                int m = g * 8 + r;
                float v = elu1(acc[ti][r] + b3[col]);
                atomicAdd(&s_msg[m][o], s_x[m][i] * v);   // ds_add_f32
            }
        }
    }
    __syncthreads();

    for (int idx = tid; idx < 16 * 64; idx += 128) {
        int m = idx >> 6, o = idx & 63;
        gatomAdd(&agg[(long)s_dst[m] * 64 + o], s_msg[m][o]);
    }
}

// ---------------------------------------------------------------------------
// NNConv node update: out[:,:,0] = x @ root_w + agg + root_b ; one wave = 16 nodes
__global__ __launch_bounds__(128) void k_node0(
    const float* __restrict__ x, const float* __restrict__ agg,
    const __bf16* __restrict__ prw, const float* __restrict__ rb,
    float* __restrict__ out) {
    int wave = threadIdx.x >> 5, lane = threadIdx.x & 31;
    int tile = blockIdx.x * 4 + wave;
    if (tile >= N_NODES / 16) return;
    long n0 = (long)tile * 16;
    int g = lane >> 4, n = lane & 15;
    v16bf a = load_a_gl(x, n0, IN_DIM, 0, IN_DIM, lane);
#pragma unroll
    for (int t = 0; t < 4; ++t) {
        v8f acc = {};
        acc = wmma_bf16(a, load_b_pk(prw + t * 512, lane), acc);
        int col = t * 16 + n;
#pragma unroll
        for (int r = 0; r < 8; ++r) {
            long node = n0 + g * 8 + r;
            float v = acc[r] + agg[node * 64 + col] + rb[col];
            out[(node * 64 + col) * 4 + 0] = v;   // [N,64,4] slot 0
        }
    }
}

// ---------------------------------------------------------------------------
__global__ void k_prep(const float* __restrict__ out, int l,
                       float* __restrict__ xelu, float* __restrict__ nbr) {
    int i = blockIdx.x * blockDim.x + threadIdx.x;
    if (i < N_NODES * 64) {
        xelu[i] = elu1(out[(long)i * 4 + l]);
        nbr[i] = 0.f;
    }
}

__global__ void k_scatter(const int* __restrict__ ei, const float* __restrict__ xelu,
                          float* __restrict__ nbr) {
    long i = (long)blockIdx.x * blockDim.x + threadIdx.x;
    if (i < (long)N_EDGES * 64) {
        int e = (int)(i >> 6), d = (int)(i & 63);
        long s = ei[e], t = ei[N_EDGES + e];
        gatomAdd(&nbr[t * 64 + d], xelu[s * 64 + d]);
    }
}

// ---------------------------------------------------------------------------
// Fused GIN MLP: h0 = xelu + nbr ; 64 -> 256 -> 256 -> 64 ; 16 nodes / workgroup
__global__ __launch_bounds__(128) void k_gin(
    const float* __restrict__ xelu, const float* __restrict__ nbr,
    const __bf16* __restrict__ w1p, const float* __restrict__ b1,
    const __bf16* __restrict__ w2p, const float* __restrict__ b2,
    const __bf16* __restrict__ w3p, const float* __restrict__ b3,
    float* __restrict__ out, int l) {
    __shared__ __bf16 s_h0[16][64];
    __shared__ __bf16 s_h1[16][256];
    __shared__ __bf16 s_h2[16][256];
    int tid = threadIdx.x, wave = tid >> 5, lane = tid & 31;
    long n0 = (long)blockIdx.x * 16;
    int g = lane >> 4, n = lane & 15;

    for (int i = tid; i < 16 * 64; i += 128) {
        int m = i >> 6, d = i & 63;
        long idx = (n0 + m) * 64 + d;
        s_h0[m][d] = (__bf16)(xelu[idx] + nbr[idx]);
    }
    __syncthreads();

    // GEMM1: [16,64] @ [64,256]   (Nt = 16, Kt = 2)
    {
        v16bf a0 = load_a_lds(&s_h0[0][0], 64, 0, lane);
        v16bf a1 = load_a_lds(&s_h0[0][0], 64, 32, lane);
        for (int t = wave; t < 16; t += 4) {
            v8f acc = {};
            acc = wmma_bf16(a0, load_b_pk(w1p + (0 * 16 + t) * 512, lane), acc);
            acc = wmma_bf16(a1, load_b_pk(w1p + (1 * 16 + t) * 512, lane), acc);
            int col = t * 16 + n;
#pragma unroll
            for (int r = 0; r < 8; ++r)
                s_h1[g * 8 + r][col] = (__bf16)elu1(acc[r] + b1[col]);
        }
    }
    __syncthreads();

    // GEMM2: [16,256] @ [256,256]   (Nt = 16, Kt = 8)
    {
        v16bf af[8];
#pragma unroll
        for (int ks = 0; ks < 8; ++ks) af[ks] = load_a_lds(&s_h1[0][0], 256, ks * 32, lane);
        for (int t = wave; t < 16; t += 4) {
            v8f acc = {};
#pragma unroll
            for (int ks = 0; ks < 8; ++ks)
                acc = wmma_bf16(af[ks], load_b_pk(w2p + (ks * 16 + t) * 512, lane), acc);
            int col = t * 16 + n;
#pragma unroll
            for (int r = 0; r < 8; ++r)
                s_h2[g * 8 + r][col] = (__bf16)elu1(acc[r] + b2[col]);
        }
    }
    __syncthreads();

    // GEMM3: [16,256] @ [256,64] -> out[:,:,l+1]   (Nt = 4, Kt = 8)
    {
        v16bf af[8];
#pragma unroll
        for (int ks = 0; ks < 8; ++ks) af[ks] = load_a_lds(&s_h2[0][0], 256, ks * 32, lane);
        int t = wave;  // 4 col tiles, one per wave
        v8f acc = {};
#pragma unroll
        for (int ks = 0; ks < 8; ++ks)
            acc = wmma_bf16(af[ks], load_b_pk(w3p + (ks * 4 + t) * 512, lane), acc);
        int col = t * 16 + n;
#pragma unroll
        for (int r = 0; r < 8; ++r) {
            long node = n0 + g * 8 + r;
            out[(node * 64 + col) * 4 + (l + 1)] = acc[r] + b3[col];
        }
    }
}

// ---------------------------------------------------------------------------
extern "C" void kernel_launch(void* const* d_in, const int* in_sizes, int n_in,
                              void* d_out, int out_size, void* d_ws, size_t ws_size,
                              hipStream_t stream) {
    const float* x     = (const float*)d_in[0];
    const int*   ei    = (const int*)d_in[1];     // [2, E]
    const float* eattr = (const float*)d_in[2];
    const float* ew1   = (const float*)d_in[3];
    const float* eb1   = (const float*)d_in[4];
    const float* ew2   = (const float*)d_in[5];
    const float* eb2   = (const float*)d_in[6];
    const float* ew3   = (const float*)d_in[7];
    const float* eb3   = (const float*)d_in[8];
    const float* rw    = (const float*)d_in[9];
    const float* rb    = (const float*)d_in[10];
    const float* gw1   = (const float*)d_in[11];
    const float* gb1   = (const float*)d_in[12];
    const float* gw2   = (const float*)d_in[13];
    const float* gb2   = (const float*)d_in[14];
    const float* gw3   = (const float*)d_in[15];
    const float* gb3   = (const float*)d_in[16];
    float* out  = (float*)d_out;                  // [N, 64, 4]
    float* nbr  = (float*)d_ws;                   // N*64 (agg / nbr accumulator)
    float* xelu = nbr + (size_t)N_NODES * 64;     // N*64

    // Packed bf16 weight region (fragment-major, 512 elems per 32x16 K-tile block)
    __bf16* pw1 = (__bf16*)(xelu + (size_t)N_NODES * 64);
    __bf16* pw2 = pw1 + 16 * 512;          // enn_w1: Kt=1,  Nt=16
    __bf16* pw3 = pw2 + 512 * 512;         // enn_w2: Kt=8,  Nt=64
    __bf16* prw = pw3 + 2048 * 512;        // enn_w3: Kt=32, Nt=64
    __bf16* pg1 = prw + 4 * 512;           // root:   Kt=1,  Nt=4
    __bf16* pg2 = pg1 + 3 * 32 * 512;      // gin_w1: Kt=2,  Nt=16 (x3)
    __bf16* pg3 = pg2 + 3 * 128 * 512;     // gin_w2: Kt=8,  Nt=16 (x3)
                                           // gin_w3: Kt=8,  Nt=4  (x3)

    auto pack = [&](const float* W, __bf16* P, int Kreal, int Kt, int Nt, int N) {
        int total = Kt * Nt * 512;
        k_pack<<<(total + 255) / 256, 256, 0, stream>>>(W, P, Kreal, Nt, N, total);
    };
    pack(ew1, pw1, 16, 1, 16, 256);
    pack(ew2, pw2, 256, 8, 64, 1024);
    pack(ew3, pw3, 1024, 32, 64, 1024);
    pack(rw, prw, 16, 1, 4, 64);
    for (int l = 0; l < 3; ++l) {
        pack(gw1 + (long)l * 64 * 256, pg1 + (long)l * 32 * 512, 64, 2, 16, 256);
        pack(gw2 + (long)l * 256 * 256, pg2 + (long)l * 128 * 512, 256, 8, 16, 256);
        pack(gw3 + (long)l * 256 * 64, pg3 + (long)l * 32 * 512, 256, 8, 4, 64);
    }

    const int nd64 = N_NODES * 64;
    k_zero<<<(nd64 + 255) / 256, 256, 0, stream>>>(nbr, nd64);
    k_edge<<<N_EDGES / 16, 128, 0, stream>>>(x, ei, eattr, pw1, eb1, pw2, eb2,
                                             pw3, eb3, nbr);
    k_node0<<<(N_NODES / 16 + 3) / 4, 128, 0, stream>>>(x, nbr, prw, rb, out);
    for (int l = 0; l < 3; ++l) {
        k_prep<<<(nd64 + 255) / 256, 256, 0, stream>>>(out, l, xelu, nbr);
        k_scatter<<<(int)(((long)N_EDGES * 64) / 256), 256, 0, stream>>>(ei, xelu, nbr);
        k_gin<<<N_NODES / 16, 128, 0, stream>>>(xelu, nbr,
                                                pg1 + (long)l * 32 * 512, gb1 + l * HIDDEN,
                                                pg2 + (long)l * 128 * 512, gb2 + l * HIDDEN,
                                                pg3 + (long)l * 32 * 512, gb3 + l * NODE_DIM,
                                                out, l);
    }
}
